// RelPos1dMultiHeadAttention_88218628260742
// MI455X (gfx1250) — matrix operand order repaired and checked by reading, not compiled, not run on hardware
//
#include <hip/hip_runtime.h>
#include <hip/hip_bf16.h>
#include <math.h>

typedef __bf16 bfx8  __attribute__((ext_vector_type(8)));
typedef __bf16 bfx16 __attribute__((ext_vector_type(16)));
typedef float  f32x8 __attribute__((ext_vector_type(8)));
typedef int    intx4 __attribute__((ext_vector_type(4)));

#define DMODEL 1024
#define NH 16
#define DH 64
#define BSZ 4
#define TLEN 1024
#define RLEN 2048   // 2T-1 = 2047 padded to 2048 (last row zero)

#define AS1 __attribute__((address_space(1)))
#define AS3 __attribute__((address_space(3)))

#if defined(__has_builtin)
#  if __has_builtin(__builtin_amdgcn_global_load_async_to_lds_b128)
#    define HAS_ASYNC_COPY 1
#  endif
#endif
#ifndef HAS_ASYNC_COPY
#  define HAS_ASYNC_COPY 0
#endif

// Copy 8 bf16 (16 B) from global to LDS. Async (ASYNCcnt) when available.
__device__ __forceinline__ void copy8_g2l(const __bf16* g, __bf16* l) {
#if HAS_ASYNC_COPY
  __builtin_amdgcn_global_load_async_to_lds_b128(
      (AS1 intx4*)(void*)g, (AS3 intx4*)(void*)l, 0, 0);
#else
  *(bfx8*)l = *(const bfx8*)g;
#endif
}

__device__ __forceinline__ void wait_async_copies() {
#if HAS_ASYNC_COPY
#  if __has_builtin(__builtin_amdgcn_s_wait_asynccnt)
  __builtin_amdgcn_s_wait_asynccnt(0);
#  else
  asm volatile("s_wait_asynccnt 0" ::: "memory");
#  endif
#endif
}

// ---------------------------------------------------------------------------
// WMMA helpers (CDNA5 wave32, 16x16x32 bf16 -> f32)
// ---------------------------------------------------------------------------
__device__ __forceinline__ f32x8 wmma_bf16(bfx16 a, bfx16 b, f32x8 c) {
  return __builtin_amdgcn_wmma_f32_16x16x32_bf16(false, a, false, b,
                                                 (short)0, c, false, false);
}

// A-matrix fragment (16x32 bf16), row-major source with leading dim `ld`.
// ISA layout: lane L holds row m = L%16; lanes 0-15 K={0..7,16..23},
// lanes 16-31 K={8..15,24..31}.
__device__ __forceinline__ bfx16 load_a_frag(const __bf16* A, int ld,
                                             int m0, int k0, int lane) {
  int m = m0 + (lane & 15);
  const __bf16* p = A + (size_t)m * ld + k0 + ((lane >> 4) << 3);
  bfx8 lo = *(const bfx8*)p;
  bfx8 hi = *(const bfx8*)(p + 16);
  bfx16 r;
#pragma unroll
  for (int i = 0; i < 8; ++i) { r[i] = lo[i]; r[i + 8] = hi[i]; }
  return r;
}

// B-matrix fragment (32x16 bf16), row-major [K][N] source with leading dim ld.
// ISA layout: lane L holds col n = L%16; lanes 0-15 K={0..7,16..23},
// lanes 16-31 K={8..15,24..31}.
__device__ __forceinline__ bfx16 load_b_frag(const __bf16* B, int ld,
                                             int k0, int n0, int lane) {
  const __bf16* p = B + (size_t)(k0 + ((lane >> 4) << 3)) * ld + n0 + (lane & 15);
  bfx16 r;
#pragma unroll
  for (int i = 0; i < 8; ++i) { r[i] = p[(size_t)i * ld]; r[i + 8] = p[(size_t)(i + 16) * ld]; }
  return r;
}

// ---------------------------------------------------------------------------
// f32 -> bf16 conversion
// ---------------------------------------------------------------------------
__global__ void cvt_f32_bf16(const float* __restrict__ in,
                             __bf16* __restrict__ out, int n) {
  int i = blockIdx.x * blockDim.x + threadIdx.x;
  if (i < n) out[i] = (__bf16)in[i];
}

// ---------------------------------------------------------------------------
// Relative sinusoidal encoding, bf16, [RLEN, DMODEL], row 2047 zeroed
// ---------------------------------------------------------------------------
__global__ void enc_kernel(__bf16* __restrict__ enc) {
  int i = blockIdx.x * blockDim.x + threadIdx.x;   // over RLEN * DMODEL/2
  int r = i >> 9;                                   // / (DMODEL/2)
  int j = i & 511;
  if (r >= RLEN) return;
  float s = 0.f, c = 0.f;
  if (r < 2 * TLEN - 1) {
    float pos  = (float)(TLEN - 1 - r);             // T-1 ... -(T-1)
    float invf = __expf(-((float)(2 * j) / (float)DMODEL) * 9.210340371976184f);
    float a = pos * invf;
    s = __sinf(a);
    c = __cosf(a);
  }
  enc[(size_t)r * DMODEL + 2 * j]     = (__bf16)s;
  enc[(size_t)r * DMODEL + 2 * j + 1] = (__bf16)c;
}

// ---------------------------------------------------------------------------
// bf16 GEMM: C[M,N] = A[M,K] @ W[K,N] + bias
//   mode 0: bf16 row-major out
//   mode 1: f32  row-major out
//   mode 2: bf16 head-transposed out: ((m/trows)*N + n)*trows + (m%trows)
// Block: 256 thr / 8 waves, tile 128x128, K-step 32, async LDS staging.
// ---------------------------------------------------------------------------
#define LDA_S 40
#define LDB_S 136

__global__ __launch_bounds__(256)
void gemm_bf16(const __bf16* __restrict__ A, const __bf16* __restrict__ Bw,
               const float* __restrict__ bias, void* __restrict__ Cout,
               int M, int N, int K, int mode, int trows) {
  __shared__ __bf16 As[128 * LDA_S];
  __shared__ __bf16 Bs[32 * LDB_S];

  int tid  = threadIdx.x;
  int lane = tid & 31;
  int w    = tid >> 5;
  int wm   = (w & 1) * 64;    // 2 M slices
  int wn   = (w >> 1) * 32;   // 4 N slices
  int m0   = blockIdx.y * 128;
  int n0   = blockIdx.x * 128;

  f32x8 acc[4][2];
#pragma unroll
  for (int fm = 0; fm < 4; ++fm)
#pragma unroll
    for (int fn = 0; fn < 2; ++fn) acc[fm][fn] = (f32x8){};

  int ra = tid >> 1, ca = (tid & 1) * 16;
  int rb = tid >> 3, cb = (tid & 7) * 16;

  for (int k0 = 0; k0 < K; k0 += 32) {
    // stage A 128x32 and B 32x128 directly into LDS (async on CDNA5)
    {
      const __bf16* ga = A + (size_t)(m0 + ra) * K + k0 + ca;
      copy8_g2l(ga,     &As[ra * LDA_S + ca]);
      copy8_g2l(ga + 8, &As[ra * LDA_S + ca + 8]);

      const __bf16* gb = Bw + (size_t)(k0 + rb) * N + n0 + cb;
      copy8_g2l(gb,     &Bs[rb * LDB_S + cb]);
      copy8_g2l(gb + 8, &Bs[rb * LDB_S + cb + 8]);

      if (k0 + 32 < K) {
        __builtin_prefetch(A + (size_t)(m0 + ra) * K + (k0 + 32) + ca, 0, 3);
        __builtin_prefetch(Bw + (size_t)(k0 + 32 + rb) * N + n0 + cb, 0, 3);
      }
    }
    wait_async_copies();
    __syncthreads();

    bfx16 bf0 = load_b_frag(&Bs[0], LDB_S, 0, wn, lane);
    bfx16 bf1 = load_b_frag(&Bs[0], LDB_S, 0, wn + 16, lane);
#pragma unroll
    for (int fm = 0; fm < 4; ++fm) {
      bfx16 af = load_a_frag(&As[0], LDA_S, wm + fm * 16, 0, lane);
      acc[fm][0] = wmma_bf16(af, bf0, acc[fm][0]);
      acc[fm][1] = wmma_bf16(af, bf1, acc[fm][1]);
    }
    __syncthreads();
  }

  // epilogue: C layout -> lane holds col n = lane&15, rows m = v + 8*(lane>>4)
  int lh = lane >> 4, lc = lane & 15;
#pragma unroll
  for (int fm = 0; fm < 4; ++fm)
#pragma unroll
    for (int fn = 0; fn < 2; ++fn)
#pragma unroll
      for (int v = 0; v < 8; ++v) {
        int m = m0 + wm + fm * 16 + v + 8 * lh;
        int n = n0 + wn + fn * 16 + lc;
        float val = acc[fm][fn][v] + bias[n];
        if (mode == 0) {
          ((__bf16*)Cout)[(size_t)m * N + n] = (__bf16)val;
        } else if (mode == 1) {
          ((float*)Cout)[(size_t)m * N + n] = val;
        } else {
          size_t addr = ((size_t)(m / trows) * N + n) * (size_t)trows + (m % trows);
          ((__bf16*)Cout)[addr] = (__bf16)val;
        }
      }
}

// ---------------------------------------------------------------------------
// Fused relative-position flash attention.
// grid (T/128, H, B), 256 thr / 8 waves; each wave owns 16 queries.
// K and V 64x64 tiles are cooperatively staged in LDS once per k-step
// (async global->LDS), shared by all 8 waves.
// Q: [B*T, D] bf16 ; Kt: [B,H,dh,T] bf16 ; V: [B*T, D] bf16 ;
// Et: [H,dh,RLEN] bf16 ; O out: [B*T, D] bf16.
// ---------------------------------------------------------------------------
#define LDKV 72   // padded LDS stride (halves) for 64-wide tiles

__global__ __launch_bounds__(256)
void attn_kernel(const __bf16* __restrict__ Qb, const __bf16* __restrict__ Ktb,
                 const __bf16* __restrict__ Vb, const __bf16* __restrict__ Etb,
                 __bf16* __restrict__ Ob) {
  __shared__ __bf16 Ks[64 * LDKV];      // [d][token]  (head-transposed K tile)
  __shared__ __bf16 Vs[64 * LDKV];      // [token][d]
  __shared__ float  Mlds[8][16 * 32];   // per-wave skew scratch
  __shared__ __bf16 Plds[8][16 * 64];   // per-wave P staging

  int tid  = threadIdx.x;
  int lane = tid & 31;
  int w    = tid >> 5;
  int lh   = lane >> 4, lc = lane & 15;
  int b = blockIdx.z, h = blockIdx.y;
  int q0 = blockIdx.x * 128 + w * 16;

  const __bf16* Qh  = Qb  + (size_t)b * TLEN * DMODEL + h * DH;
  const __bf16* Kth = Ktb + (size_t)(b * NH + h) * DH * TLEN;
  const __bf16* Vh  = Vb  + (size_t)b * TLEN * DMODEL + h * DH;
  const __bf16* Eth = Etb + (size_t)h * DH * RLEN;

  bfx16 aq[2];
  aq[0] = load_a_frag(Qh, DMODEL, q0, 0, lane);
  aq[1] = load_a_frag(Qh, DMODEL, q0, 32, lane);

  f32x8 o[4];
#pragma unroll
  for (int jn = 0; jn < 4; ++jn) o[jn] = (f32x8){};
  float rm[8], rl[8];
#pragma unroll
  for (int v = 0; v < 8; ++v) { rm[v] = -1e30f; rl[v] = 0.f; }

  int tr = tid >> 2, tc = (tid & 3) * 16;   // cooperative stage coords

  for (int k0 = 0; k0 < TLEN; k0 += 64) {
    // ---- stage K (d-major) and V (token-major) 64x64 tiles into LDS
    {
      const __bf16* gk = Kth + (size_t)tr * TLEN + k0 + tc;
      copy8_g2l(gk,     &Ks[tr * LDKV + tc]);
      copy8_g2l(gk + 8, &Ks[tr * LDKV + tc + 8]);
      const __bf16* gv = Vh + (size_t)(k0 + tr) * DMODEL + tc;
      copy8_g2l(gv,     &Vs[tr * LDKV + tc]);
      copy8_g2l(gv + 8, &Vs[tr * LDKV + tc + 8]);
      if (k0 + 64 < TLEN) {
        __builtin_prefetch(Kth + (size_t)tr * TLEN + k0 + 64 + tc, 0, 3);
        __builtin_prefetch(Vh + (size_t)(k0 + 64 + tr) * DMODEL + tc, 0, 3);
      }
    }
    wait_async_copies();
    __syncthreads();

    f32x8 s[4];
#pragma unroll
    for (int j = 0; j < 4; ++j) s[j] = (f32x8){};

    // ---- content scores: S = Q @ K^T (B frags from LDS K tile)
#pragma unroll
    for (int j = 0; j < 4; ++j) {
      bfx16 bk0 = load_b_frag(&Ks[0], LDKV, 0,  j * 16, lane);
      s[j] = wmma_bf16(aq[0], bk0, s[j]);
      bfx16 bk1 = load_b_frag(&Ks[0], LDKV, 32, j * 16, lane);
      s[j] = wmma_bf16(aq[1], bk1, s[j]);
    }

    // ---- relative scores with on-the-fly skewing:
    // S_E[q,k] = (Q @ Et)[q, k-q+T-1]; per 16x16 tile the band starts at
    // rbase = 1008 + k0 + j*16 - q0 (always in [0,2016], reads pad row 2047)
#pragma unroll
    for (int j = 0; j < 4; ++j) {
      int rbase = 1008 + k0 + j * 16 - q0;
      f32x8 e0 = (f32x8){}, e1 = (f32x8){};
#pragma unroll
      for (int dd = 0; dd < 2; ++dd) {
        bfx16 be0 = load_b_frag(Eth, RLEN, dd * 32, rbase,      lane);
        e0 = wmma_bf16(aq[dd], be0, e0);
        bfx16 be1 = load_b_frag(Eth, RLEN, dd * 32, rbase + 16, lane);
        e1 = wmma_bf16(aq[dd], be1, e1);
      }
#pragma unroll
      for (int v = 0; v < 8; ++v) {
        int m = v + 8 * lh;
        Mlds[w][m * 32 + lc]      = e0[v];
        Mlds[w][m * 32 + 16 + lc] = e1[v];
      }
      __syncthreads();
#pragma unroll
      for (int v = 0; v < 8; ++v) {
        int m = v + 8 * lh;
        int c = lc - m + 15;                 // in [0,30]
        s[j][v] += Mlds[w][m * 32 + c];
      }
      __syncthreads();
    }

    // ---- online softmax (scale 1/sqrt(dh) = 0.125)
#pragma unroll
    for (int j = 0; j < 4; ++j)
#pragma unroll
      for (int v = 0; v < 8; ++v) s[j][v] *= 0.125f;

#pragma unroll
    for (int v = 0; v < 8; ++v) {
      float mx = fmaxf(fmaxf(s[0][v], s[1][v]), fmaxf(s[2][v], s[3][v]));
#pragma unroll
      for (int off = 1; off < 16; off <<= 1)
        mx = fmaxf(mx, __shfl_xor(mx, off, 32));
      float mnew  = fmaxf(rm[v], mx);
      float alpha = __expf(rm[v] - mnew);
      rm[v] = mnew;
      float sum = 0.f;
#pragma unroll
      for (int j = 0; j < 4; ++j) {
        float p = __expf(s[j][v] - mnew);
        s[j][v] = p;
        sum += p;
      }
#pragma unroll
      for (int off = 1; off < 16; off <<= 1)
        sum += __shfl_xor(sum, off, 32);
      rl[v] = rl[v] * alpha + sum;
#pragma unroll
      for (int jn = 0; jn < 4; ++jn) o[jn][v] *= alpha;
    }

    // ---- P (C layout) -> LDS -> A fragments, then O += P @ V (V from LDS)
#pragma unroll
    for (int j = 0; j < 4; ++j)
#pragma unroll
      for (int v = 0; v < 8; ++v) {
        int m = v + 8 * lh;
        Plds[w][m * 64 + j * 16 + lc] = (__bf16)s[j][v];
      }
    __syncthreads();
    bfx16 ap0 = load_a_frag(&Plds[w][0], 64, 0, 0,  lane);
    bfx16 ap1 = load_a_frag(&Plds[w][0], 64, 0, 32, lane);
#pragma unroll
    for (int jn = 0; jn < 4; ++jn) {
      bfx16 bv0 = load_b_frag(&Vs[0], LDKV, 0,  jn * 16, lane);
      o[jn] = wmma_bf16(ap0, bv0, o[jn]);
      bfx16 bv1 = load_b_frag(&Vs[0], LDKV, 32, jn * 16, lane);
      o[jn] = wmma_bf16(ap1, bv1, o[jn]);
    }
    __syncthreads();   // protects Ks/Vs/Plds before next iteration's staging
  }

  // ---- normalize and store O (bf16, [b*T + q][h*64 + d])
#pragma unroll
  for (int jn = 0; jn < 4; ++jn)
#pragma unroll
    for (int v = 0; v < 8; ++v) {
      int m = v + 8 * lh;
      int d = jn * 16 + lc;
      float val = o[jn][v] / rl[v];
      Ob[(size_t)(b * TLEN + q0 + m) * DMODEL + h * DH + d] = (__bf16)val;
    }
}

// ---------------------------------------------------------------------------
// Launch
// ---------------------------------------------------------------------------
extern "C" void kernel_launch(void* const* d_in, const int* in_sizes, int n_in,
                              void* d_out, int out_size, void* d_ws, size_t ws_size,
                              hipStream_t stream) {
  const float* x    = (const float*)d_in[0];
  const float* Wq   = (const float*)d_in[1];
  const float* bq   = (const float*)d_in[2];
  const float* Wk   = (const float*)d_in[3];
  const float* bk   = (const float*)d_in[4];
  const float* Wv   = (const float*)d_in[5];
  const float* bv   = (const float*)d_in[6];
  const float* Wpos = (const float*)d_in[7];
  const float* bpos = (const float*)d_in[8];
  const float* Wo   = (const float*)d_in[9];
  const float* bo   = (const float*)d_in[10];

  char* w = (char*)d_ws;
  const size_t MB = 1024 * 1024;
  __bf16* xb   = (__bf16*)(w + 0 * MB);    // 8 MB
  __bf16* wqb  = (__bf16*)(w + 8 * MB);    // 2 MB
  __bf16* wkb  = (__bf16*)(w + 10 * MB);
  __bf16* wvb  = (__bf16*)(w + 12 * MB);
  __bf16* wpb  = (__bf16*)(w + 14 * MB);
  __bf16* wob  = (__bf16*)(w + 16 * MB);
  __bf16* encb = (__bf16*)(w + 18 * MB);   // 4 MB
  __bf16* Qb   = (__bf16*)(w + 22 * MB);   // 8 MB
  __bf16* Ktb  = (__bf16*)(w + 30 * MB);   // 8 MB
  __bf16* Vb   = (__bf16*)(w + 38 * MB);   // 8 MB
  __bf16* Etb  = (__bf16*)(w + 46 * MB);   // 4 MB
  __bf16* Ob   = (__bf16*)(w + 50 * MB);   // 8 MB  (total 58 MB)

  const int nx = BSZ * TLEN * DMODEL;      // 4M
  const int nw = DMODEL * DMODEL;          // 1M
  cvt_f32_bf16<<<(nx + 255) / 256, 256, 0, stream>>>(x,    xb,  nx);
  cvt_f32_bf16<<<(nw + 255) / 256, 256, 0, stream>>>(Wq,   wqb, nw);
  cvt_f32_bf16<<<(nw + 255) / 256, 256, 0, stream>>>(Wk,   wkb, nw);
  cvt_f32_bf16<<<(nw + 255) / 256, 256, 0, stream>>>(Wv,   wvb, nw);
  cvt_f32_bf16<<<(nw + 255) / 256, 256, 0, stream>>>(Wpos, wpb, nw);
  cvt_f32_bf16<<<(nw + 255) / 256, 256, 0, stream>>>(Wo,   wob, nw);

  enc_kernel<<<(RLEN * (DMODEL / 2)) / 256, 256, 0, stream>>>(encb);

  dim3 gproj(DMODEL / 128, (BSZ * TLEN) / 128);   // (8, 32)
  gemm_bf16<<<gproj, 256, 0, stream>>>(xb, wqb, bq, Qb,  BSZ * TLEN, DMODEL, DMODEL, 0, 1);
  gemm_bf16<<<gproj, 256, 0, stream>>>(xb, wkb, bk, Ktb, BSZ * TLEN, DMODEL, DMODEL, 2, TLEN);
  gemm_bf16<<<gproj, 256, 0, stream>>>(xb, wvb, bv, Vb,  BSZ * TLEN, DMODEL, DMODEL, 0, 1);

  dim3 ge(DMODEL / 128, RLEN / 128);              // (8, 16)
  gemm_bf16<<<ge, 256, 0, stream>>>(encb, wpb, bpos, Etb, RLEN, DMODEL, DMODEL, 2, RLEN);

  dim3 ga(TLEN / 128, NH, BSZ);
  attn_kernel<<<ga, 256, 0, stream>>>(Qb, Ktb, Vb, Etb, Ob);

  gemm_bf16<<<gproj, 256, 0, stream>>>(Ob, wob, bo, d_out, BSZ * TLEN, DMODEL, DMODEL, 1, 1);
}
